// Attention_39728447488068
// MI455X (gfx1250) — matrix operand order repaired and checked by reading, not compiled, not run on hardware
//
#include <hip/hip_runtime.h>
#include <hip/hip_bf16.h>
#include <stddef.h>

typedef __bf16 bf16;
typedef __attribute__((ext_vector_type(16))) __bf16 v16bf;
typedef __attribute__((ext_vector_type(8)))  __bf16 v8bf;
typedef __attribute__((ext_vector_type(8)))  float  v8f;

#define DEV __device__ __forceinline__

DEV int lane_id() { return threadIdx.x & 31; }

union BF16x16 { v16bf v; v8bf h[2]; };

DEV v8f zero_v8f() {
  v8f z;
  #pragma unroll
  for (int i = 0; i < 8; ++i) z[i] = 0.f;
  return z;
}

// A fragment (16x32, bf16), memory row-major [16][ld], K contiguous.
// Lane l: row = l%16, half h=l/16; elems e<8 -> K=8h+e ; e>=8 -> K=8h+16+(e-8)
DEV v16bf load_frag_a(const bf16* base, int ld) {
  const int l = lane_id(), r = l & 15, hh = l >> 4;
  const bf16* p = base + (size_t)r * ld + 8 * hh;
  BF16x16 u;
  u.h[0] = *(const v8bf*)(p);
  u.h[1] = *(const v8bf*)(p + 16);
  return u.v;
}

// B fragment (32x16, bf16) where memory holds [N][K] row-major (K contiguous):
// element (K = 8h + e for e<8, K = 16 + 8h + (e-8) for e>=8, N = l%16).
DEV v16bf load_frag_b(const bf16* base, int ld) {
  const int l = lane_id(), r = l & 15, hh = l >> 4;
  const bf16* p = base + (size_t)r * ld + 8 * hh;
  BF16x16 u;
  u.h[0] = *(const v8bf*)(p);
  u.h[1] = *(const v8bf*)(p + 16);
  return u.v;
}

DEV v8f wmma_bf16(v16bf a, v16bf b, v8f c) {
  return __builtin_amdgcn_wmma_f32_16x16x32_bf16(false, a, false, b,
                                                 (short)0, c, false, false);
}

// ---------------------------------------------------------------------------
// fp32 -> bf16 conversion
// ---------------------------------------------------------------------------
__global__ __launch_bounds__(256)
void cvt_f32_bf16_kernel(const float* __restrict__ src, bf16* __restrict__ dst,
                         size_t n) {
  size_t i = (size_t)blockIdx.x * blockDim.x + threadIdx.x;
  size_t stride = (size_t)gridDim.x * blockDim.x;
  for (; i < n; i += stride) dst[i] = (bf16)src[i];
}

// ---------------------------------------------------------------------------
// C[M,N] = A[M,K] * W[N,K]^T  (both bf16 row-major, K contiguous -> "NT" GEMM)
// block = 256 threads = 8 waves in 2(M) x 4(N); block tile 128x128,
// per-wave 64x32 = 4x2 WMMA tiles, K stepped by 32.
// ---------------------------------------------------------------------------
template <bool OUT_BF16>
__global__ __launch_bounds__(256)
void gemm_nt_kernel(const bf16* __restrict__ A, const bf16* __restrict__ W,
                    void* __restrict__ Cout, int M, int N, int K) {
  const int w  = threadIdx.x >> 5;
  const int wm = w >> 2, wn = w & 3;
  const int m0 = blockIdx.y * 128 + wm * 64;
  const int n0 = blockIdx.x * 128 + wn * 32;

  v8f acc[4][2];
  #pragma unroll
  for (int i = 0; i < 4; ++i)
    #pragma unroll
    for (int j = 0; j < 2; ++j) acc[i][j] = zero_v8f();

  for (int k = 0; k < K; k += 32) {
    if (k + 256 < K) {  // CDNA5 global_prefetch_b8 path
      __builtin_prefetch(A + (size_t)m0 * K + k + 256, 0, 1);
      __builtin_prefetch(W + (size_t)n0 * K + k + 256, 0, 1);
    }
    v16bf a[4], bb[2];
    #pragma unroll
    for (int i = 0; i < 4; ++i)
      a[i] = load_frag_a(A + (size_t)(m0 + 16 * i) * K + k, K);
    #pragma unroll
    for (int j = 0; j < 2; ++j)
      bb[j] = load_frag_b(W + (size_t)(n0 + 16 * j) * K + k, K);
    #pragma unroll
    for (int i = 0; i < 4; ++i)
      #pragma unroll
      for (int j = 0; j < 2; ++j)
        acc[i][j] = wmma_bf16(a[i], bb[j], acc[i][j]);
  }

  const int l = lane_id(), n = l & 15, hh = l >> 4;
  #pragma unroll
  for (int i = 0; i < 4; ++i)
    #pragma unroll
    for (int j = 0; j < 2; ++j)
      #pragma unroll
      for (int v = 0; v < 8; ++v) {
        const int row = m0 + 16 * i + v + 8 * hh;
        const int col = n0 + 16 * j + n;
        const float val = acc[i][j][v];
        if constexpr (OUT_BF16)
          ((bf16*)Cout)[(size_t)row * N + col] = (bf16)val;
        else
          ((float*)Cout)[(size_t)row * N + col] = val;
      }
}

// ---------------------------------------------------------------------------
// RoPE: src [b*s, nheads*128] -> dst [b, nheads, s, 128], 64 threads = pairs
// ---------------------------------------------------------------------------
__global__ __launch_bounds__(64)
void rope_kernel(const bf16* __restrict__ src, bf16* __restrict__ dst,
                 const float* __restrict__ cosf, const float* __restrict__ sinf,
                 int nheads) {
  const int S = 2048;
  int idx = blockIdx.x;
  const int h = idx % nheads; idx /= nheads;
  const int s = idx % S;
  const int b = idx / S;
  const int t = threadIdx.x;  // pair index 0..63
  const size_t so = ((size_t)(b * S + s) * nheads + h) * 128;
  const float xr = (float)src[so + 2 * t];
  const float xi = (float)src[so + 2 * t + 1];
  const float c  = cosf[s * 64 + t];
  const float sn = sinf[s * 64 + t];
  const size_t doff = ((size_t)(b * nheads + h) * S + s) * 128;
  dst[doff + 2 * t]     = (bf16)(xr * c - xi * sn);
  dst[doff + 2 * t + 1] = (bf16)(xr * sn + xi * c);
}

// V: [b*s, 4*128] -> [b, 4, s, 128]
__global__ __launch_bounds__(128)
void vtrans_kernel(const bf16* __restrict__ src, bf16* __restrict__ dst) {
  const int S = 2048, NKV = 4;
  int idx = blockIdx.x;
  const int h = idx % NKV; idx /= NKV;
  const int s = idx % S;
  const int b = idx / S;
  const int t = threadIdx.x;
  dst[((size_t)(b * NKV + h) * S + s) * 128 + t] =
      src[((size_t)(b * S + s) * NKV + h) * 128 + t];
}

// ---------------------------------------------------------------------------
// Causal flash attention. Grid: (S/64, B*NH). Block: 128 threads = 4 waves,
// each wave owns 16 query rows. Q/K in [b,head,s,128] bf16; V in
// [b,kvhead,s,128] bf16. Output written to ao[b*s, 2048] bf16.
// ---------------------------------------------------------------------------
#define FA_VPAD 72
#define FA_PPAD 72

__global__ __launch_bounds__(128)
void flash_attn_kernel(const bf16* __restrict__ Q, const bf16* __restrict__ Km,
                       const bf16* __restrict__ Vt, bf16* __restrict__ Out) {
  __shared__ bf16 lds_vT[128 * FA_VPAD];      // V tile transposed: [d][kpos]
  __shared__ bf16 lds_p[4][16 * FA_PPAD];     // per-wave P staging

  const int S = 2048, NH = 16, NKV = 4;
  const int bh  = blockIdx.y;
  const int b   = bh / NH, h = bh % NH, hkv = h >> 2;  // head -> kv = h/4
  const int q0  = blockIdx.x * 64;
  const int w   = threadIdx.x >> 5;
  const int l   = lane_id(), n = l & 15, hh = l >> 4;

  // Q fragments for this wave's 16 rows, all 128 d (4 K-steps of 32)
  const bf16* qbase = Q + ((size_t)(b * NH + h) * S + (q0 + 16 * w)) * 128;
  v16bf aq[4];
  #pragma unroll
  for (int ks = 0; ks < 4; ++ks) aq[ks] = load_frag_a(qbase + 32 * ks, 128);

  v8f o[8];
  float m_i[8], l_i[8];
  #pragma unroll
  for (int f = 0; f < 8; ++f) o[f] = zero_v8f();
  #pragma unroll
  for (int v = 0; v < 8; ++v) { m_i[v] = -3.0e38f; l_i[v] = 0.f; }

  const bf16* kbase = Km + ((size_t)(b * NKV + hkv) * S) * 128;
  const bf16* vbase = Vt + ((size_t)(b * NKV + hkv) * S) * 128;
  const float scale = 0.088388347648318447f;  // 1/sqrt(128)

  for (int kt = 0; kt <= (int)blockIdx.x; ++kt) {
    const int k0 = kt * 64;

    // cooperative transpose of 64x128 V tile into LDS as [d][kpos]
    {
      const int d = threadIdx.x;                 // 0..127
      const bf16* vsrc = vbase + (size_t)k0 * 128;
      #pragma unroll 4
      for (int kp = 0; kp < 64; ++kp)
        lds_vT[d * FA_VPAD + kp] = vsrc[(size_t)kp * 128 + d];
    }
    __syncthreads();

    // S = Q K^T : 16 rows x 64 keys, K-dim = 128
    v8f sacc[4];
    #pragma unroll
    for (int j = 0; j < 4; ++j) sacc[j] = zero_v8f();
    #pragma unroll
    for (int ks = 0; ks < 4; ++ks) {
      #pragma unroll
      for (int j = 0; j < 4; ++j) {
        v16bf bk =
            load_frag_b(kbase + (size_t)(k0 + 16 * j) * 128 + 32 * ks, 128);
        sacc[j] = wmma_bf16(aq[ks], bk, sacc[j]);
      }
    }

    // scale + causal mask (only the diagonal tile needs it)
    const bool diag = (kt == (int)blockIdx.x);
    #pragma unroll
    for (int j = 0; j < 4; ++j)
      #pragma unroll
      for (int v = 0; v < 8; ++v) {
        float s = sacc[j][v] * scale;
        if (diag) {
          const int colg = k0 + 16 * j + n;
          const int rowg = q0 + 16 * w + v + 8 * hh;
          if (colg > rowg) s = -3.0e38f;
        }
        sacc[j][v] = s;
      }

    // online softmax: row reductions across the 16 lanes holding one row
    #pragma unroll
    for (int v = 0; v < 8; ++v) {
      float t = sacc[0][v];
      t = fmaxf(t, sacc[1][v]);
      t = fmaxf(t, sacc[2][v]);
      t = fmaxf(t, sacc[3][v]);
      for (int off = 1; off < 16; off <<= 1)
        t = fmaxf(t, __shfl_xor(t, off, 32));
      const float mn = fmaxf(m_i[v], t);
      const float alpha = __expf(m_i[v] - mn);
      m_i[v] = mn;
      float rs = 0.f;
      #pragma unroll
      for (int j = 0; j < 4; ++j) {
        const float p = __expf(sacc[j][v] - mn);
        sacc[j][v] = p;
        rs += p;
      }
      for (int off = 1; off < 16; off <<= 1) rs += __shfl_xor(rs, off, 32);
      l_i[v] = l_i[v] * alpha + rs;
      #pragma unroll
      for (int f = 0; f < 8; ++f) o[f][v] *= alpha;
    }

    // stage P (bf16) to per-wave LDS, re-read in A-fragment layout
    bf16* pw = &lds_p[w][0];
    #pragma unroll
    for (int j = 0; j < 4; ++j)
      #pragma unroll
      for (int v = 0; v < 8; ++v)
        pw[(v + 8 * hh) * FA_PPAD + 16 * j + n] = (bf16)sacc[j][v];

    // O += P V : K-dim = 64 keys (2 steps of 32), N-dim = 128 d (8 frags)
    #pragma unroll
    for (int ks2 = 0; ks2 < 2; ++ks2) {
      v16bf pa = load_frag_a(pw + 32 * ks2, FA_PPAD);
      #pragma unroll
      for (int f = 0; f < 8; ++f) {
        v16bf vb = load_frag_b(&lds_vT[(16 * f) * FA_VPAD + 32 * ks2], FA_VPAD);
        o[f] = wmma_bf16(pa, vb, o[f]);
      }
    }
    __syncthreads();
  }

  // normalize + write [b*s, 16*128]
  #pragma unroll
  for (int f = 0; f < 8; ++f)
    #pragma unroll
    for (int v = 0; v < 8; ++v) {
      const int rowg = q0 + 16 * w + v + 8 * hh;
      const int col  = h * 128 + 16 * f + n;
      Out[((size_t)(b * S) + rowg) * 2048 + col] = (bf16)(o[f][v] / l_i[v]);
    }
}

// ---------------------------------------------------------------------------
// Host launch
// ---------------------------------------------------------------------------
extern "C" void kernel_launch(void* const* d_in, const int* in_sizes, int n_in,
                              void* d_out, int out_size, void* d_ws,
                              size_t ws_size, hipStream_t stream) {
  (void)in_sizes; (void)n_in; (void)out_size; (void)ws_size;
  const int B = 2, S = 2048, D = 2048, NH = 16, NKV = 4, KV = 512;
  const int M = B * S;  // 4096

  const float* x  = (const float*)d_in[0];
  const float* fc = (const float*)d_in[1];
  const float* fs = (const float*)d_in[2];
  const float* wq = (const float*)d_in[3];
  const float* wk = (const float*)d_in[4];
  const float* wv = (const float*)d_in[5];
  const float* wo = (const float*)d_in[6];

  char* ws = (char*)d_ws;
  size_t off = 0;
  auto take = [&](size_t bytes) -> char* {
    char* p = ws + off;
    off += (bytes + 255) & ~(size_t)255;
    return p;
  };
  bf16* xb  = (bf16*)take((size_t)M * D * 2);
  bf16* wqb = (bf16*)take((size_t)D * D * 2);
  bf16* wkb = (bf16*)take((size_t)KV * D * 2);
  bf16* wvb = (bf16*)take((size_t)KV * D * 2);
  bf16* wob = (bf16*)take((size_t)D * D * 2);
  bf16* qb  = (bf16*)take((size_t)M * D * 2);
  bf16* kb  = (bf16*)take((size_t)M * KV * 2);
  bf16* vb  = (bf16*)take((size_t)M * KV * 2);
  bf16* qr  = (bf16*)take((size_t)M * D * 2);
  bf16* kr  = (bf16*)take((size_t)M * KV * 2);
  bf16* vt  = (bf16*)take((size_t)M * KV * 2);
  bf16* ao  = qb;  // qb fully consumed by rope before attention writes ao

  // fp32 -> bf16
  cvt_f32_bf16_kernel<<<1024, 256, 0, stream>>>(x,  xb,  (size_t)M * D);
  cvt_f32_bf16_kernel<<<1024, 256, 0, stream>>>(wq, wqb, (size_t)D * D);
  cvt_f32_bf16_kernel<<<512,  256, 0, stream>>>(wk, wkb, (size_t)KV * D);
  cvt_f32_bf16_kernel<<<512,  256, 0, stream>>>(wv, wvb, (size_t)KV * D);
  cvt_f32_bf16_kernel<<<1024, 256, 0, stream>>>(wo, wob, (size_t)D * D);

  // QKV projections (WMMA)
  gemm_nt_kernel<true><<<dim3(D / 128,  M / 128), 256, 0, stream>>>(xb, wqb, qb, M, D,  D);
  gemm_nt_kernel<true><<<dim3(KV / 128, M / 128), 256, 0, stream>>>(xb, wkb, kb, M, KV, D);
  gemm_nt_kernel<true><<<dim3(KV / 128, M / 128), 256, 0, stream>>>(xb, wvb, vb, M, KV, D);

  // RoPE + layout transforms
  rope_kernel<<<B * S * NH,  64, 0, stream>>>(qb, qr, fc, fs, NH);
  rope_kernel<<<B * S * NKV, 64, 0, stream>>>(kb, kr, fc, fs, NKV);
  vtrans_kernel<<<B * S * NKV, 128, 0, stream>>>(vb, vt);

  // causal flash attention (WMMA)
  flash_attn_kernel<<<dim3(S / 64, B * NH), 128, 0, stream>>>(qr, kr, vt, ao);

  // output projection, fp32 out (WMMA)
  gemm_nt_kernel<false><<<dim3(D / 128, M / 128), 256, 0, stream>>>(
      ao, wob, d_out, M, D, D);
}
